// SimpleSSMLayer_26362509263276
// MI455X (gfx1250) — matrix operand order repaired and checked by reading, not compiled, not run on hardware
//
#include <hip/hip_runtime.h>
#include <hip/hip_bf16.h>
#include <math.h>

// SimpleSSMLayer for MI455X (gfx1250, wave32, WMMA + async global->LDS copies).
// Pipeline: [bf16 pre-convert x/Wi/Wo] [dv-mean]
//   -> [GEMM1 bf16-WMMA, async-LDS double-buffered: xz = x@Wi^T+bi]
//   -> [dt: softplus(x_in[:,:32]@Wdt^T+bdt)]
//   -> [fused conv4+SiLU+SSM scan, bf16 output]
//   -> [GEMM2 bf16-WMMA: y@Wo^T+bo+x*mean(Dv)]

#define D_MODEL 1024
#define D_INNER 2048
#define D_STATE 16
#define DT_RANK 32
#define N_XZ    (2 * D_INNER)   // 4096
#define T_SEQ   2048

typedef __attribute__((ext_vector_type(16))) __bf16 v16bf;
typedef __attribute__((ext_vector_type(8)))  __bf16 bf16x8;
typedef __attribute__((ext_vector_type(8)))  float  v8f;

union FragU { v16bf v; bf16x8 h[2]; };

__device__ __forceinline__ __bf16 f2bf(float f) {
  unsigned u = __builtin_bit_cast(unsigned, f);
  u += 0x7FFFu + ((u >> 16) & 1u);                 // round-to-nearest-even
  unsigned short hs = (unsigned short)(u >> 16);
  return __builtin_bit_cast(__bf16, hs);
}

__device__ __forceinline__ bf16x8 cvt8(float4 a, float4 b) {
  bf16x8 r;
  r[0] = f2bf(a.x); r[1] = f2bf(a.y); r[2] = f2bf(a.z); r[3] = f2bf(a.w);
  r[4] = f2bf(b.x); r[5] = f2bf(b.y); r[6] = f2bf(b.z); r[7] = f2bf(b.w);
  return r;
}

__device__ __forceinline__ float sigmoidf_(float x) { return 1.0f / (1.0f + expf(-x)); }

// ---------------------------------------------------------------------------
// f32 -> bf16 bulk convert (8 elements per thread, b128 stores)
// ---------------------------------------------------------------------------
__global__ __launch_bounds__(256)
void cvt_bf16_kernel(const float* __restrict__ src, __bf16* __restrict__ dst, int n8) {
  const int i = blockIdx.x * blockDim.x + threadIdx.x;
  if (i >= n8) return;
  const float4* s = (const float4*)src;
  float4 a = s[2 * i], b = s[2 * i + 1];
  *(bf16x8*)&dst[8 * (size_t)i] = cvt8(a, b);
}

// ---------------------------------------------------------------------------
// Dv mean reduction (2048 -> 1 scalar)
// ---------------------------------------------------------------------------
__global__ void dvmean_kernel(const float* __restrict__ Dv, float* __restrict__ out) {
  __shared__ float s[256];
  int tid = threadIdx.x;
  float a = 0.f;
#pragma unroll
  for (int i = 0; i < D_INNER / 256; ++i) a += Dv[tid + i * 256];
  s[tid] = a;
  __syncthreads();
  for (int off = 128; off > 0; off >>= 1) {
    if (tid < off) s[tid] += s[tid + off];
    __syncthreads();
  }
  if (tid == 0) out[0] = s[0] / (float)D_INNER;
}

// ---------------------------------------------------------------------------
// Tiled bf16 WMMA GEMM:  C[M,N] = A[M,K]*W[N,K]^T + bias[N]  (+ resid*dv)
// Both operands bf16 in global memory. LDS staging uses CDNA5
// GLOBAL_LOAD_ASYNC_TO_LDS_B128 with double buffering (ASYNCcnt-tracked),
// overlapping the next k-chunk's copies with the current chunk's 8 WMMAs.
// Block: 256 thr = 8 waves, tile 128x128x32; wave grid 2(M) x 4(N).
// ---------------------------------------------------------------------------
template <bool RESID>
__global__ __launch_bounds__(256)
void gemm_bf16_wmma_kernel(const __bf16* __restrict__ A,
                           const __bf16* __restrict__ W,
                           const float* __restrict__ bias,
                           const float* __restrict__ resid,
                           const float* __restrict__ dvm,
                           float* __restrict__ C,
                           int M, int N, int K) {
  constexpr int BM = 128, BN = 128, BK = 32, LDS_STRIDE = BK + 8; // 40 bf16 = 80B rows
  __shared__ alignas(16) __bf16 sA[2][BM][LDS_STRIDE];
  __shared__ alignas(16) __bf16 sB[2][BN][LDS_STRIDE];

  const int tid  = threadIdx.x;
  const int lane = tid & 31;
  const int wave = tid >> 5;
  const int wm   = wave & 1;       // 0..1  -> 64-row half
  const int wn   = wave >> 1;      // 0..3  -> 32-col slice
  const int m0   = blockIdx.y * BM;
  const int n0   = blockIdx.x * BN;

  const int frow = tid >> 1;            // 0..127 : staging row
  const int fcol = (tid & 1) * 16;      // 0 / 16 : staging col (bf16 units)

  v8f acc[4][2];
#pragma unroll
  for (int i = 0; i < 4; ++i)
#pragma unroll
    for (int j = 0; j < 2; ++j) acc[i][j] = (v8f)(0.0f);

  const int g  = lane >> 4;    // 0/1 : K-group
  const int lr = lane & 15;    // row/col within 16-tile

  // issue 4 async b128 global->LDS copies for one k-chunk (A & W tiles)
  auto stage = [&](int buf, int k0) {
    unsigned ldsA = (unsigned)(uintptr_t)&sA[buf][frow][fcol];
    const __bf16* gA = &A[(size_t)(m0 + frow) * K + k0 + fcol];
    asm volatile("global_load_async_to_lds_b128 %0, %1, off"
                 :: "v"(ldsA), "v"(gA) : "memory");
    asm volatile("global_load_async_to_lds_b128 %0, %1, off"
                 :: "v"(ldsA + 16u), "v"(gA + 8) : "memory");
    unsigned ldsB = (unsigned)(uintptr_t)&sB[buf][frow][fcol];
    const __bf16* gW = &W[(size_t)(n0 + frow) * K + k0 + fcol];
    asm volatile("global_load_async_to_lds_b128 %0, %1, off"
                 :: "v"(ldsB), "v"(gW) : "memory");
    asm volatile("global_load_async_to_lds_b128 %0, %1, off"
                 :: "v"(ldsB + 16u), "v"(gW + 8) : "memory");
  };

  const int nk = K / BK;
  stage(0, 0);
  for (int i = 0; i < nk; ++i) {
    const int cur = i & 1;
    if (i + 1 < nk) {
      stage(cur ^ 1, (i + 1) * BK);                    // overlap with compute below
      asm volatile("s_wait_asynccnt 0x4" ::: "memory"); // current chunk's copies done
    } else {
      asm volatile("s_wait_asynccnt 0x0" ::: "memory");
    }
    __syncthreads();   // all waves' copies for `cur` landed in LDS

    // ---- build fragments per ISA 16-bit layouts ----
    FragU a[4], b[2];
#pragma unroll
    for (int mi = 0; mi < 4; ++mi) {
      const __bf16* row = &sA[cur][wm * 64 + mi * 16 + lr][0];
      a[mi].h[0] = *(const bf16x8*)(row + g * 8);        // K = 8g .. 8g+7
      a[mi].h[1] = *(const bf16x8*)(row + 16 + g * 8);   // K = 16+8g .. 16+8g+7
    }
#pragma unroll
    for (int nj = 0; nj < 2; ++nj) {
      const __bf16* row = &sB[cur][wn * 32 + nj * 16 + lr][0];
      b[nj].h[0] = *(const bf16x8*)(row + g * 8);        // K = 8g .. 8g+7
      b[nj].h[1] = *(const bf16x8*)(row + 16 + g * 8);   // K = 16+8g .. 16+8g+7
    }

#pragma unroll
    for (int mi = 0; mi < 4; ++mi)
#pragma unroll
      for (int nj = 0; nj < 2; ++nj)
        acc[mi][nj] = __builtin_amdgcn_wmma_f32_16x16x32_bf16(
            /*neg_a=*/false, a[mi].v, /*neg_b=*/false, b[nj].v,
            /*c_mod=*/(short)0, acc[mi][nj], /*reuse_a=*/false, /*reuse_b=*/false);

    __syncthreads();   // WAR: next stage() may overwrite the other buffer
  }

  // ---- epilogue: C layout  M = r + 8*(lane>=16), N = lane&15 ----
  const float dv = RESID ? dvm[0] : 0.0f;
#pragma unroll
  for (int mi = 0; mi < 4; ++mi) {
#pragma unroll
    for (int nj = 0; nj < 2; ++nj) {
      const int gcol    = n0 + wn * 32 + nj * 16 + lr;
      const int rowbase = m0 + wm * 64 + mi * 16 + g * 8;
      const float bs = bias[gcol];
      size_t idx = (size_t)rowbase * N + gcol;
#pragma unroll
      for (int r = 0; r < 8; ++r) {
        float v = acc[mi][nj][r] + bs;
        if (RESID) v += resid[idx] * dv;
        C[idx] = v;
        idx += (size_t)N;
      }
    }
  }
}

// ---------------------------------------------------------------------------
// dt = softplus(x_in[:, :32] @ Wdt^T + bdt)   (K=32, VALU kernel)
// ---------------------------------------------------------------------------
__global__ __launch_bounds__(256)
void dt_kernel(const float* __restrict__ xz, const float* __restrict__ Wdt,
               const float* __restrict__ bdt, float* __restrict__ dtb, int M) {
  __shared__ float sx[DT_RANK];
  const int m = blockIdx.y;
  const int n = blockIdx.x * 256 + threadIdx.x;
  if (threadIdx.x < DT_RANK) sx[threadIdx.x] = xz[(size_t)m * N_XZ + threadIdx.x];
  __syncthreads();

  float acc = bdt[n];
  const float4* w = (const float4*)&Wdt[(size_t)n * DT_RANK];
#pragma unroll
  for (int i = 0; i < DT_RANK / 4; ++i) {
    float4 wv = w[i];
    acc += wv.x * sx[4 * i] + wv.y * sx[4 * i + 1] + wv.z * sx[4 * i + 2] + wv.w * sx[4 * i + 3];
  }
  float sp = (acc > 20.0f) ? acc : log1pf(expf(acc));
  dtb[(size_t)m * D_INNER + n] = sp;
}

// ---------------------------------------------------------------------------
// Fused causal depthwise conv(4) + SiLU + sequential SSM scan + z-gate.
// One lane per (b, d, n): n in 0..15, 16-lane shfl_xor reduction for y.
// ---------------------------------------------------------------------------
__global__ __launch_bounds__(256)
void scan_kernel(const float* __restrict__ xz, const float* __restrict__ dtb,
                 const float* __restrict__ conv_w, const float* __restrict__ conv_b,
                 const float* __restrict__ A_log, const float* __restrict__ Bm,
                 const float* __restrict__ Cm, __bf16* __restrict__ ybf,
                 int n_channels /* = B * D_INNER */) {
  const int gid = blockIdx.x * blockDim.x + threadIdx.x;
  const int c   = gid >> 4;           // channel index (b,d)
  const int n   = gid & 15;           // state index
  if (c >= n_channels) return;
  const int b = c / D_INNER;
  const int d = c % D_INNER;

  const float Av = -expf(A_log[d * D_STATE + n]);
  const float Bv = Bm[d * D_STATE + n];
  const float Cv = Cm[d * D_STATE + n];
  const float cw0 = conv_w[d * 4 + 0], cw1 = conv_w[d * 4 + 1];
  const float cw2 = conv_w[d * 4 + 2], cw3 = conv_w[d * 4 + 3];
  const float cb  = conv_b[d];

  float x0 = 0.f, x1 = 0.f, x2 = 0.f;   // causal pad window
  float h  = 0.f;

  const size_t rbase = (size_t)b * T_SEQ;
  for (int t = 0; t < T_SEQ; ++t) {
    const size_t row = rbase + t;
    const float xin = xz[row * N_XZ + d];
    float conv = x0 * cw0 + x1 * cw1 + x2 * cw2 + xin * cw3 + cb;
    x0 = x1; x1 = x2; x2 = xin;
    const float u   = conv * sigmoidf_(conv);          // SiLU
    const float dtv = dtb[row * D_INNER + d];

    h = h * __expf(dtv * Av) + dtv * Bv * u;

    float p = h * Cv;                                   // reduce over 16 states
    p += __shfl_xor(p, 1);
    p += __shfl_xor(p, 2);
    p += __shfl_xor(p, 4);
    p += __shfl_xor(p, 8);
    if (n == 0) {
      const float z = xz[row * N_XZ + D_INNER + d];
      const float y = p * (z * sigmoidf_(z));           // gate with SiLU(z)
      ybf[row * D_INNER + d] = f2bf(y);
    }
  }
}

// ---------------------------------------------------------------------------
extern "C" void kernel_launch(void* const* d_in, const int* in_sizes, int n_in,
                              void* d_out, int out_size, void* d_ws, size_t ws_size,
                              hipStream_t stream) {
  const float* x      = (const float*)d_in[0];
  const float* Wi     = (const float*)d_in[1];
  const float* bi     = (const float*)d_in[2];
  const float* conv_w = (const float*)d_in[3];
  const float* conv_b = (const float*)d_in[4];
  const float* Wdt    = (const float*)d_in[5];
  const float* bdt    = (const float*)d_in[6];
  const float* A_log  = (const float*)d_in[7];
  const float* Bm     = (const float*)d_in[8];
  const float* Cm     = (const float*)d_in[9];
  const float* Dv     = (const float*)d_in[10];
  const float* Wo     = (const float*)d_in[11];
  const float* bo     = (const float*)d_in[12];
  float* out = (float*)d_out;

  const int M = in_sizes[0] / D_MODEL;       // B*T = 8192
  const int B = M / T_SEQ;

  // workspace layout
  char* ws = (char*)d_ws;
  float*  xz   = (float*)ws;                    ws += (size_t)M * N_XZ * sizeof(float);
  float*  dtb  = (float*)ws;                    ws += (size_t)M * D_INNER * sizeof(float);
  __bf16* ybf  = (__bf16*)ws;                   ws += (size_t)M * D_INNER * sizeof(__bf16);
  __bf16* x_bf = (__bf16*)ws;                   ws += (size_t)M * D_MODEL * sizeof(__bf16);
  __bf16* Wi_bf = (__bf16*)ws;                  ws += (size_t)N_XZ * D_MODEL * sizeof(__bf16);
  __bf16* Wo_bf = (__bf16*)ws;                  ws += (size_t)D_MODEL * D_INNER * sizeof(__bf16);
  float*  dvm  = (float*)ws;

  // one-shot bf16 conversions (operands then L2-resident as bf16)
  {
    int n8 = (M * D_MODEL) / 8;
    cvt_bf16_kernel<<<(n8 + 255) / 256, 256, 0, stream>>>(x, x_bf, n8);
    n8 = (N_XZ * D_MODEL) / 8;
    cvt_bf16_kernel<<<(n8 + 255) / 256, 256, 0, stream>>>(Wi, Wi_bf, n8);
    n8 = (D_MODEL * D_INNER) / 8;
    cvt_bf16_kernel<<<(n8 + 255) / 256, 256, 0, stream>>>(Wo, Wo_bf, n8);
  }
  dvmean_kernel<<<1, 256, 0, stream>>>(Dv, dvm);

  // GEMM1: xz = x @ Wi^T + bi   (M x 4096, K=1024)
  gemm_bf16_wmma_kernel<false><<<dim3(N_XZ / 128, M / 128), 256, 0, stream>>>(
      x_bf, Wi_bf, bi, nullptr, nullptr, xz, M, N_XZ, D_MODEL);

  // dt = softplus(x_in[:, :32] @ Wdt^T + bdt)
  dt_kernel<<<dim3(D_INNER / 256, M), 256, 0, stream>>>(xz, Wdt, bdt, dtb, M);

  // fused conv + SiLU + scan + z-gate -> ybf (bf16)
  const int n_channels = B * D_INNER;
  scan_kernel<<<(n_channels * D_STATE + 255) / 256, 256, 0, stream>>>(
      xz, dtb, conv_w, conv_b, A_log, Bm, Cm, ybf, n_channels);

  // GEMM2: out = ybf @ Wo^T + bo + x * mean(Dv)   (M x 1024, K=2048)
  gemm_bf16_wmma_kernel<true><<<dim3(D_MODEL / 128, M / 128), 256, 0, stream>>>(
      ybf, Wo_bf, bo, x, dvm, out, M, D_MODEL, D_INNER);
}
